// HyperCHARM_10677288698630
// MI455X (gfx1250) — matrix operand order repaired and checked by reading, not compiled, not run on hardware
//
#include <hip/hip_runtime.h>
#include <hip/hip_bf16.h>

#define N_NODES 100000
#define N_EDGES 800000
#define N_HE    50000
#define DIM     128
#define HEDIM   64
#define NGRAPH  64

typedef __attribute__((ext_vector_type(16))) _Float16 v16h;
typedef __attribute__((ext_vector_type(8)))  float    v8f;

// ---------------------------------------------------------------------------
// WMMA helpers (layouts per CDNA5 ISA 7.12.2, wave32)
// ---------------------------------------------------------------------------
__device__ inline v8f wmma_f16(v16h a, v16h b, v8f c) {
  return __builtin_amdgcn_wmma_f32_16x16x32_f16(false, a, false, b, (short)0, c, false, false);
}

// A fragment 16x32 (f16) from an fp32 row split in two segments:
//   K < lena          -> a[K]
//   lena <= K < +lenb -> b[K-lena] * s
//   else              -> 0
__device__ inline v16h loadA_split(const float* a, int lena, const float* b, float s,
                                   int lenb, int kbase, int l16, int hi) {
  v16h v;
#pragma unroll
  for (int i = 0; i < 16; ++i) {
    int K = kbase + hi * 16 + i;
    float f = 0.0f;
    if (K < lena) f = a[K];
    else if (K < lena + lenb) f = b[K - lena] * s;
    v[i] = (_Float16)f;
  }
  return v;
}

// A fragment from a row-major f16 LDS tile [16][stride]
__device__ inline v16h loadA_lds(const _Float16* t, int stride, int kbase, int l16, int hi) {
  const _Float16* row = t + l16 * stride;
  v16h v;
#pragma unroll
  for (int i = 0; i < 16; ++i) {
    int K = kbase + hi * 16 + i;
    v[i] = row[K];
  }
  return v;
}

// B fragment 32x16 from row-major f16 weight W[Ncols][Kp] (B[k][n] = W[n][k]).
// element i -> K = kbase + hi*16 + i, N = nbase + l16  (contiguous 32B per lane)
__device__ inline v16h loadB_rm(const _Float16* W, int Kp, int nbase, int kbase, int l16, int hi) {
  const _Float16* p = W + (size_t)(nbase + l16) * Kp + kbase + hi * 16;
  v16h v;
#pragma unroll
  for (int i = 0; i < 16; ++i) v[i] = p[i];
  return v;
}

__device__ inline void initAccBias(v8f acc[8], const float* bias, int l16) {
#pragma unroll
  for (int nt = 0; nt < 8; ++nt) {
    float bb = bias[nt * 16 + l16];
    v8f c;
#pragma unroll
    for (int r = 0; r < 8; ++r) c[r] = bb;
    acc[nt] = c;
  }
}

// Row LayerNorm on C fragments: row M = r + hi*8 lives in one 16-lane half.
__device__ inline void rowLN(v8f acc[8], const float* gg, const float* bb, int l16) {
  float gv[8], bv[8];
#pragma unroll
  for (int nt = 0; nt < 8; ++nt) { int c = nt * 16 + l16; gv[nt] = gg[c]; bv[nt] = bb[c]; }
#pragma unroll
  for (int r = 0; r < 8; ++r) {
    float s = 0.f, s2 = 0.f;
#pragma unroll
    for (int nt = 0; nt < 8; ++nt) { float x = acc[nt][r]; s += x; s2 += x * x; }
#pragma unroll
    for (int m = 1; m < 16; m <<= 1) { s += __shfl_xor(s, m, 32); s2 += __shfl_xor(s2, m, 32); }
    float mean = s * (1.0f / 128.0f);
    float var  = s2 * (1.0f / 128.0f) - mean * mean;
    float rs   = rsqrtf(var + 1e-5f);
#pragma unroll
    for (int nt = 0; nt < 8; ++nt) acc[nt][r] = (acc[nt][r] - mean) * rs * gv[nt] + bv[nt];
  }
}

__device__ inline unsigned fenc(float f) {
  unsigned u = __float_as_uint(f);
  return (u & 0x80000000u) ? ~u : (u | 0x80000000u);
}
__device__ inline float fdec(unsigned u) {
  return (u & 0x80000000u) ? __uint_as_float(u & 0x7fffffffu) : __uint_as_float(~u);
}

// ---------------------------------------------------------------------------
// Small utility kernels
// ---------------------------------------------------------------------------
__global__ void fill_f32(float* p, float v, size_t n) {
  size_t i = (size_t)blockIdx.x * blockDim.x + threadIdx.x;
  if (i < n) p[i] = v;
}
__global__ void init_smax(unsigned* p, int n) {
  int i = blockIdx.x * blockDim.x + threadIdx.x;
  if (i < n) p[i] = fenc(-3.0e38f);
}
__global__ void cvt_w(const float* src, _Float16* dst, int ks, int kd) {
  int r = blockIdx.x;
  for (int k = threadIdx.x; k < kd; k += blockDim.x)
    dst[(size_t)r * kd + k] = (k < ks) ? (_Float16)src[(size_t)r * ks + k] : (_Float16)0.f;
}
__global__ void deg_accum(const int* ids, float* deg, int e) {
  int i = blockIdx.x * blockDim.x + threadIdx.x;
  if (i < e) atomicAdd(&deg[ids[i]], 1.0f);
}
__global__ void recip_eps(const float* s, float* d, int n) {
  int i = blockIdx.x * blockDim.x + threadIdx.x;
  if (i < n) d[i] = 1.0f / (s[i] + 1e-6f);
}
__global__ void q_kernel(const float* seed, const float* w, const float* b, float* q) {
  int j = threadIdx.x;
  float s = b[j];
  for (int k = 0; k < 128; ++k) s += seed[k] * w[j * 128 + k];
  q[j] = s * 0.17677669529663687f;  // 1/sqrt(32)
}

// ---------------------------------------------------------------------------
// h = relu(x @ Wip.T + b)
// ---------------------------------------------------------------------------
__global__ __launch_bounds__(128) void in_proj_kernel(const float* x, const _Float16* W,
                                                      const float* bias, float* h) {
  int lane = threadIdx.x & 31, w = threadIdx.x >> 5;
  int l16 = lane & 15, hi = (lane >> 4) & 1;
  int rowbase = blockIdx.x * 64 + w * 16;
  int rA = rowbase + l16; if (rA >= N_NODES) rA = N_NODES - 1;
  const float* xr = x + (size_t)rA * DIM;
  v8f acc[8]; initAccBias(acc, bias, l16);
#pragma unroll
  for (int kc = 0; kc < 4; ++kc) {
    v16h a = loadA_split(xr, 128, nullptr, 0.f, 0, kc * 32, l16, hi);
#pragma unroll
    for (int nt = 0; nt < 8; ++nt)
      acc[nt] = wmma_f16(a, loadB_rm(W, 128, nt * 16, kc * 32, l16, hi), acc[nt]);
  }
#pragma unroll
  for (int r = 0; r < 8; ++r) {
    int n = rowbase + hi * 8 + r;
    if (n < N_NODES) {
#pragma unroll
      for (int nt = 0; nt < 8; ++nt) {
        float v = acc[nt][r];
        h[(size_t)n * DIM + nt * 16 + l16] = v > 0.f ? v : 0.f;
      }
    }
  }
}

// ---------------------------------------------------------------------------
// Fused n2e MLP: msg = relu(LN(concat(h[nid],mark[hid]) @ W1.T + b1)) @ W2.T + b2
// scatter: aggE[hid] += msg
// ---------------------------------------------------------------------------
__global__ __launch_bounds__(128) void n2e_kernel(
    const float* h, const float* he_mark, const int* node_ids, const int* he_ids,
    const _Float16* W1, const _Float16* W2,
    const float* b1, const float* gln, const float* bln, const float* b2, float* aggE) {
  __shared__ _Float16 tbuf[4][16][132];
  int lane = threadIdx.x & 31, w = threadIdx.x >> 5;
  int l16 = lane & 15, hi = (lane >> 4) & 1;
  int rowbase = blockIdx.x * 64 + w * 16;
  int eA = rowbase + l16; if (eA >= N_EDGES) eA = N_EDGES - 1;
  const float* hrow = h + (size_t)node_ids[eA] * DIM;
  const float* mrow = he_mark + (size_t)he_ids[eA] * 2;

  v8f acc[8]; initAccBias(acc, b1, l16);
#pragma unroll
  for (int kc = 0; kc < 5; ++kc) {           // K = 130 padded to 160
    v16h a = loadA_split(hrow, 128, mrow, 1.0f, 2, kc * 32, l16, hi);
#pragma unroll
    for (int nt = 0; nt < 8; ++nt)
      acc[nt] = wmma_f16(a, loadB_rm(W1, 160, nt * 16, kc * 32, l16, hi), acc[nt]);
  }
  rowLN(acc, gln, bln, l16);
#pragma unroll
  for (int nt = 0; nt < 8; ++nt)
#pragma unroll
    for (int r = 0; r < 8; ++r) {
      float v = acc[nt][r]; v = v > 0.f ? v : 0.f;
      tbuf[w][hi * 8 + r][nt * 16 + l16] = (_Float16)v;
    }
  __syncthreads();
  v8f acc2[8]; initAccBias(acc2, b2, l16);
#pragma unroll
  for (int kc = 0; kc < 4; ++kc) {
    v16h a = loadA_lds(&tbuf[w][0][0], 132, kc * 32, l16, hi);
#pragma unroll
    for (int nt = 0; nt < 8; ++nt)
      acc2[nt] = wmma_f16(a, loadB_rm(W2, 128, nt * 16, kc * 32, l16, hi), acc2[nt]);
  }
#pragma unroll
  for (int r = 0; r < 8; ++r) {
    int e = rowbase + hi * 8 + r;
    if (e < N_EDGES) {
      int he = he_ids[e];
#pragma unroll
      for (int nt = 0; nt < 8; ++nt)
        atomicAdd(&aggE[(size_t)he * DIM + nt * 16 + l16], acc2[nt][r]);
    }
  }
}

// ---------------------------------------------------------------------------
// Fused e2n MLP: inc = relu(relu(LN(concat(attr[hid], aggE[hid]*inv) @ W1.T + b1)) @ W2.T + b2)
// scatter: aggN[nid] += inc
// ---------------------------------------------------------------------------
__global__ __launch_bounds__(128) void e2n_kernel(
    const float* he_attr, const float* aggE, const float* invcnt,
    const int* node_ids, const int* he_ids,
    const _Float16* W1, const _Float16* W2,
    const float* b1, const float* gln, const float* bln, const float* b2, float* aggN) {
  __shared__ _Float16 tbuf[4][16][132];
  int lane = threadIdx.x & 31, w = threadIdx.x >> 5;
  int l16 = lane & 15, hi = (lane >> 4) & 1;
  int rowbase = blockIdx.x * 64 + w * 16;
  int eA = rowbase + l16; if (eA >= N_EDGES) eA = N_EDGES - 1;
  int heA = he_ids[eA];
  const float* arow = he_attr + (size_t)heA * HEDIM;
  const float* grow = aggE + (size_t)heA * DIM;
  float inv = invcnt[heA];

  v8f acc[8]; initAccBias(acc, b1, l16);
#pragma unroll
  for (int kc = 0; kc < 6; ++kc) {           // K = 64 + 128 = 192
    v16h a = loadA_split(arow, 64, grow, inv, 128, kc * 32, l16, hi);
#pragma unroll
    for (int nt = 0; nt < 8; ++nt)
      acc[nt] = wmma_f16(a, loadB_rm(W1, 192, nt * 16, kc * 32, l16, hi), acc[nt]);
  }
  rowLN(acc, gln, bln, l16);
#pragma unroll
  for (int nt = 0; nt < 8; ++nt)
#pragma unroll
    for (int r = 0; r < 8; ++r) {
      float v = acc[nt][r]; v = v > 0.f ? v : 0.f;
      tbuf[w][hi * 8 + r][nt * 16 + l16] = (_Float16)v;
    }
  __syncthreads();
  v8f acc2[8]; initAccBias(acc2, b2, l16);
#pragma unroll
  for (int kc = 0; kc < 4; ++kc) {
    v16h a = loadA_lds(&tbuf[w][0][0], 132, kc * 32, l16, hi);
#pragma unroll
    for (int nt = 0; nt < 8; ++nt)
      acc2[nt] = wmma_f16(a, loadB_rm(W2, 128, nt * 16, kc * 32, l16, hi), acc2[nt]);
  }
#pragma unroll
  for (int r = 0; r < 8; ++r) {
    int e = rowbase + hi * 8 + r;
    if (e < N_EDGES) {
      int nid = node_ids[e];
#pragma unroll
      for (int nt = 0; nt < 8; ++nt) {
        float v = acc2[nt][r]; v = v > 0.f ? v : 0.f;
        atomicAdd(&aggN[(size_t)nid * DIM + nt * 16 + l16], v);
      }
    }
  }
}

// ---------------------------------------------------------------------------
// h += LN(aggN * invdeg)    (one warp per node row)
// ---------------------------------------------------------------------------
__global__ __launch_bounds__(128) void node_update(float* h, const float* aggN,
                                                   const float* invdeg,
                                                   const float* g, const float* b) {
  int lane = threadIdx.x & 31, wid = threadIdx.x >> 5;
  int n = blockIdx.x * 4 + wid;
  if (n >= N_NODES) return;
  float inv = invdeg[n];
  float x[4], s = 0.f, s2 = 0.f;
#pragma unroll
  for (int i = 0; i < 4; ++i) {
    int c = lane + i * 32;
    x[i] = aggN[(size_t)n * DIM + c] * inv;
    s += x[i]; s2 += x[i] * x[i];
  }
#pragma unroll
  for (int m = 1; m < 32; m <<= 1) { s += __shfl_xor(s, m, 32); s2 += __shfl_xor(s2, m, 32); }
  float mean = s * (1.0f / 128.0f);
  float var  = s2 * (1.0f / 128.0f) - mean * mean;
  float rs   = rsqrtf(var + 1e-5f);
#pragma unroll
  for (int i = 0; i < 4; ++i) {
    int c = lane + i * 32;
    h[(size_t)n * DIM + c] += (x[i] - mean) * rs * g[c] + b[c];
  }
}

// ---------------------------------------------------------------------------
// K/V projections + masked scores + segment-max (encoded atomicMax)
// ---------------------------------------------------------------------------
__global__ __launch_bounds__(128) void kv_kernel(
    const float* h, const _Float16* Wk, const _Float16* Wv,
    const float* bk, const float* bv, const float* qbuf,
    const int* batch, const int* node_pos, const int* resp,
    float* scores, unsigned* smax_enc, float* vv) {
  int lane = threadIdx.x & 31, w = threadIdx.x >> 5;
  int l16 = lane & 15, hi = (lane >> 4) & 1;
  int rowbase = blockIdx.x * 64 + w * 16;
  int rA = rowbase + l16; if (rA >= N_NODES) rA = N_NODES - 1;
  const float* hrow = h + (size_t)rA * DIM;

  v8f acc[8]; initAccBias(acc, bk, l16);
#pragma unroll
  for (int kc = 0; kc < 4; ++kc) {
    v16h a = loadA_split(hrow, 128, nullptr, 0.f, 0, kc * 32, l16, hi);
#pragma unroll
    for (int nt = 0; nt < 8; ++nt)
      acc[nt] = wmma_f16(a, loadB_rm(Wk, 128, nt * 16, kc * 32, l16, hi), acc[nt]);
  }
  float qv[8];
#pragma unroll
  for (int nt = 0; nt < 8; ++nt) qv[nt] = qbuf[(nt >> 1) * 32 + (nt & 1) * 16 + l16];
#pragma unroll
  for (int r = 0; r < 8; ++r) {
    float sh[4] = {0.f, 0.f, 0.f, 0.f};
#pragma unroll
    for (int nt = 0; nt < 8; ++nt) sh[nt >> 1] += acc[nt][r] * qv[nt];
#pragma unroll
    for (int m = 1; m < 16; m <<= 1) {
#pragma unroll
      for (int hd = 0; hd < 4; ++hd) sh[hd] += __shfl_xor(sh[hd], m, 32);
    }
    int n = rowbase + hi * 8 + r;
    if (n < N_NODES && l16 == 0) {
      int g = batch[n];
      bool msk = node_pos[n] >= resp[g];
#pragma unroll
      for (int hd = 0; hd < 4; ++hd) {
        float s = msk ? sh[hd] : -1e30f;
        scores[(size_t)n * 4 + hd] = s;
        atomicMax(&smax_enc[g * 4 + hd], fenc(s));
      }
    }
  }
  // V projection
  v8f accv[8]; initAccBias(accv, bv, l16);
#pragma unroll
  for (int kc = 0; kc < 4; ++kc) {
    v16h a = loadA_split(hrow, 128, nullptr, 0.f, 0, kc * 32, l16, hi);
#pragma unroll
    for (int nt = 0; nt < 8; ++nt)
      accv[nt] = wmma_f16(a, loadB_rm(Wv, 128, nt * 16, kc * 32, l16, hi), accv[nt]);
  }
#pragma unroll
  for (int r = 0; r < 8; ++r) {
    int n = rowbase + hi * 8 + r;
    if (n < N_NODES) {
#pragma unroll
      for (int nt = 0; nt < 8; ++nt)
        vv[(size_t)n * DIM + nt * 16 + l16] = accv[nt][r];
    }
  }
}

// ---------------------------------------------------------------------------
// exp/weighted-sum accumulation per node
// ---------------------------------------------------------------------------
__global__ __launch_bounds__(128) void p2_kernel(
    const float* scores, const unsigned* smax_enc, const float* vv,
    const int* batch, const int* node_pos, const int* resp,
    float* ctx, float* denom, float* counts) {
  int n = blockIdx.x, t = threadIdx.x;
  int g = batch[n];
  float mf = (node_pos[n] >= resp[g]) ? 1.f : 0.f;
  int hd = t >> 5;
  float sm = fdec(smax_enc[g * 4 + hd]);
  if (sm <= -1e29f) sm = 0.f;
  float s = scores[(size_t)n * 4 + hd];
  float e = expf(s - sm) * mf;
  atomicAdd(&ctx[g * DIM + t], e * vv[(size_t)n * DIM + t]);
  if ((t & 31) == 0) atomicAdd(&denom[g * 4 + hd], e);
  if (t == 0) atomicAdd(&counts[g], mf);
}

// ---------------------------------------------------------------------------
// Per-graph tail: out-proj + LN + FFN + LN + pred head (tiny, VALU)
// ---------------------------------------------------------------------------
__device__ inline float blockSum(float v, float* sred) {
  int t = threadIdx.x;
  sred[t] = v; __syncthreads();
  for (int s = 64; s > 0; s >>= 1) { if (t < s) sred[t] += sred[t + s]; __syncthreads(); }
  float r = sred[0]; __syncthreads();
  return r;
}

__global__ __launch_bounds__(128) void p3_kernel(
    const float* ctx, const float* denom, const float* counts, const float* seed,
    const float* mow, const float* mob,
    const float* f1w, const float* f1b, const float* f2w, const float* f2b,
    const float* n1g, const float* n1b, const float* n2g, const float* n2b,
    const float* p1w, const float* p1b, const float* p2w, const float* p2b,
    float* out) {
  __shared__ float sctx[128], so1[128], shid[512], shg[128], sred[128], sh2[64];
  int g = blockIdx.x, t = threadIdx.x;
  float dnm = denom[g * 4 + (t >> 5)];
  if (!(dnm > 0.f)) dnm = 1.f;
  sctx[t] = ctx[g * DIM + t] / dnm;
  __syncthreads();
  float att = mob[t];
  for (int k = 0; k < 128; ++k) att += sctx[k] * mow[t * 128 + k];
  float x1 = att + seed[t];
  float s = blockSum(x1, sred), s2 = blockSum(x1 * x1, sred);
  float mean = s / 128.f, var = s2 / 128.f - mean * mean, rs = rsqrtf(var + 1e-5f);
  float o1 = (x1 - mean) * rs * n1g[t] + n1b[t];
  so1[t] = o1; __syncthreads();
  for (int j = t; j < 512; j += 128) {
    float hv = f1b[j];
    for (int k = 0; k < 128; ++k) hv += so1[k] * f1w[j * 128 + k];
    shid[j] = hv > 0.f ? hv : 0.f;
  }
  __syncthreads();
  float o2 = f2b[t];
  for (int k = 0; k < 512; ++k) o2 += shid[k] * f2w[t * 512 + k];
  float x2 = o1 + o2;
  s = blockSum(x2, sred); s2 = blockSum(x2 * x2, sred);
  mean = s / 128.f; var = s2 / 128.f - mean * mean; rs = rsqrtf(var + 1e-5f);
  float pma = (x2 - mean) * rs * n2g[t] + n2b[t];
  float cnt = counts[g];
  if (!(cnt > 0.f)) pma = 0.f;
  float hg = pma / sqrtf(fmaxf(cnt, 1.f));
  shg[t] = hg; __syncthreads();
  if (t < 64) {
    float hv = p1b[t];
    for (int k = 0; k < 128; ++k) hv += shg[k] * p1w[t * 128 + k];
    sh2[t] = hv > 0.f ? hv : 0.f;
  }
  __syncthreads();
  if (t == 0) {
    float p = p2b[0];
    for (int j = 0; j < 64; ++j) p += sh2[j] * p2w[j];
    out[g] = p;
  }
}

// ---------------------------------------------------------------------------
extern "C" void kernel_launch(void* const* d_in, const int* in_sizes, int n_in,
                              void* d_out, int out_size, void* d_ws, size_t ws_size,
                              hipStream_t stream) {
  (void)in_sizes; (void)n_in; (void)out_size; (void)ws_size;
  const float* x         = (const float*)d_in[0];
  const float* he_attr   = (const float*)d_in[1];
  const float* he_mark   = (const float*)d_in[2];
  const float* he_count  = (const float*)d_in[3];
  const float* in_proj_w = (const float*)d_in[4];
  const float* in_proj_b = (const float*)d_in[5];
  const float* n2e_w1    = (const float*)d_in[6];
  const float* n2e_b1    = (const float*)d_in[7];
  const float* n2e_g     = (const float*)d_in[8];
  const float* n2e_bn    = (const float*)d_in[9];
  const float* n2e_w2    = (const float*)d_in[10];
  const float* n2e_b2    = (const float*)d_in[11];
  const float* e2n_w1    = (const float*)d_in[12];
  const float* e2n_b1    = (const float*)d_in[13];
  const float* e2n_g     = (const float*)d_in[14];
  const float* e2n_bn    = (const float*)d_in[15];
  const float* e2n_w2    = (const float*)d_in[16];
  const float* e2n_b2    = (const float*)d_in[17];
  const float* lnout_g   = (const float*)d_in[18];
  const float* lnout_b   = (const float*)d_in[19];
  const float* seed      = (const float*)d_in[20];
  const float* mha_in_w  = (const float*)d_in[21];
  const float* mha_in_b  = (const float*)d_in[22];
  const float* mha_out_w = (const float*)d_in[23];
  const float* mha_out_b = (const float*)d_in[24];
  const float* ffn_w1    = (const float*)d_in[25];
  const float* ffn_b1    = (const float*)d_in[26];
  const float* ffn_w2    = (const float*)d_in[27];
  const float* ffn_b2    = (const float*)d_in[28];
  const float* n1_g      = (const float*)d_in[29];
  const float* n1_b      = (const float*)d_in[30];
  const float* n2_g      = (const float*)d_in[31];
  const float* n2_b      = (const float*)d_in[32];
  const float* pred_w1   = (const float*)d_in[33];
  const float* pred_b1   = (const float*)d_in[34];
  const float* pred_w2   = (const float*)d_in[35];
  const float* pred_b2   = (const float*)d_in[36];
  const int* node_ids    = (const int*)d_in[37];
  const int* he_ids      = (const int*)d_in[38];
  const int* batch       = (const int*)d_in[39];
  const int* node_pos    = (const int*)d_in[40];
  const int* resp        = (const int*)d_in[41];

  char* wsb = (char*)d_ws;
  size_t off = 0;
  auto alloc = [&](size_t bytes) -> char* {
    char* p = wsb + off;
    off += (bytes + 255) & ~(size_t)255;
    return p;
  };
  float* h       = (float*)alloc((size_t)N_NODES * DIM * 4);
  float* bufA    = (float*)alloc((size_t)N_NODES * DIM * 4);   // aggN, later vv
  float* aggE    = (float*)alloc((size_t)N_HE * DIM * 4);
  float* scores  = (float*)alloc((size_t)N_NODES * 4 * 4);
  float* deg     = (float*)alloc((size_t)N_NODES * 4);
  float* invdeg  = (float*)alloc((size_t)N_NODES * 4);
  float* invcnt  = (float*)alloc((size_t)N_HE * 4);
  float* qbuf    = (float*)alloc(128 * 4);
  unsigned* smaxe= (unsigned*)alloc(256 * 4);
  float* denom   = (float*)alloc(256 * 4);
  float* ctx     = (float*)alloc((size_t)NGRAPH * DIM * 4);
  float* counts  = (float*)alloc((size_t)NGRAPH * 4);
  _Float16* wip_h   = (_Float16*)alloc((size_t)128 * 128 * 2);
  _Float16* n2e_w1h = (_Float16*)alloc((size_t)2 * 128 * 160 * 2);
  _Float16* n2e_w2h = (_Float16*)alloc((size_t)2 * 128 * 128 * 2);
  _Float16* e2n_w1h = (_Float16*)alloc((size_t)2 * 128 * 192 * 2);
  _Float16* e2n_w2h = (_Float16*)alloc((size_t)2 * 128 * 128 * 2);
  _Float16* wk_h    = (_Float16*)alloc((size_t)128 * 128 * 2);
  _Float16* wv_h    = (_Float16*)alloc((size_t)128 * 128 * 2);

  // ---- prep ----
  fill_f32<<<(N_NODES + 255) / 256, 256, 0, stream>>>(deg, 0.f, N_NODES);
  deg_accum<<<(N_EDGES + 255) / 256, 256, 0, stream>>>(node_ids, deg, N_EDGES);
  recip_eps<<<(N_NODES + 255) / 256, 256, 0, stream>>>(deg, invdeg, N_NODES);
  recip_eps<<<(N_HE + 255) / 256, 256, 0, stream>>>(he_count, invcnt, N_HE);
  cvt_w<<<128, 128, 0, stream>>>(in_proj_w, wip_h, 128, 128);
  for (int l = 0; l < 2; ++l) {
    cvt_w<<<128, 128, 0, stream>>>(n2e_w1 + (size_t)l * 128 * 130, n2e_w1h + (size_t)l * 128 * 160, 130, 160);
    cvt_w<<<128, 128, 0, stream>>>(n2e_w2 + (size_t)l * 128 * 128, n2e_w2h + (size_t)l * 128 * 128, 128, 128);
    cvt_w<<<128, 128, 0, stream>>>(e2n_w1 + (size_t)l * 128 * 192, e2n_w1h + (size_t)l * 128 * 192, 192, 192);
    cvt_w<<<128, 128, 0, stream>>>(e2n_w2 + (size_t)l * 128 * 128, e2n_w2h + (size_t)l * 128 * 128, 128, 128);
  }
  cvt_w<<<128, 128, 0, stream>>>(mha_in_w + 128 * 128, wk_h, 128, 128);
  cvt_w<<<128, 128, 0, stream>>>(mha_in_w + 2 * 128 * 128, wv_h, 128, 128);

  // ---- in-projection ----
  in_proj_kernel<<<(N_NODES + 63) / 64, 128, 0, stream>>>(x, wip_h, in_proj_b, h);

  // ---- message-passing layers ----
  for (int l = 0; l < 2; ++l) {
    fill_f32<<<(int)(((size_t)N_HE * DIM + 255) / 256), 256, 0, stream>>>(aggE, 0.f, (size_t)N_HE * DIM);
    n2e_kernel<<<(N_EDGES + 63) / 64, 128, 0, stream>>>(
        h, he_mark, node_ids, he_ids,
        n2e_w1h + (size_t)l * 128 * 160, n2e_w2h + (size_t)l * 128 * 128,
        n2e_b1 + l * 128, n2e_g + l * 128, n2e_bn + l * 128, n2e_b2 + l * 128, aggE);
    fill_f32<<<(int)(((size_t)N_NODES * DIM + 255) / 256), 256, 0, stream>>>(bufA, 0.f, (size_t)N_NODES * DIM);
    e2n_kernel<<<(N_EDGES + 63) / 64, 128, 0, stream>>>(
        he_attr, aggE, invcnt, node_ids, he_ids,
        e2n_w1h + (size_t)l * 128 * 192, e2n_w2h + (size_t)l * 128 * 128,
        e2n_b1 + l * 128, e2n_g + l * 128, e2n_bn + l * 128, e2n_b2 + l * 128, bufA);
    node_update<<<(N_NODES + 3) / 4, 128, 0, stream>>>(h, bufA, invdeg,
                                                       lnout_g + l * 128, lnout_b + l * 128);
  }

  // ---- PMA pooling + head ----
  q_kernel<<<1, 128, 0, stream>>>(seed, mha_in_w, mha_in_b, qbuf);
  fill_f32<<<(NGRAPH * DIM + 255) / 256, 256, 0, stream>>>(ctx, 0.f, (size_t)NGRAPH * DIM);
  fill_f32<<<1, 256, 0, stream>>>(denom, 0.f, 256);
  fill_f32<<<1, 64, 0, stream>>>(counts, 0.f, NGRAPH);
  init_smax<<<1, 256, 0, stream>>>(smaxe, 256);
  kv_kernel<<<(N_NODES + 63) / 64, 128, 0, stream>>>(
      h, wk_h, wv_h, mha_in_b + 128, mha_in_b + 256, qbuf,
      batch, node_pos, resp, scores, smaxe, bufA);
  p2_kernel<<<N_NODES, 128, 0, stream>>>(scores, smaxe, bufA, batch, node_pos, resp,
                                         ctx, denom, counts);
  p3_kernel<<<NGRAPH, 128, 0, stream>>>(ctx, denom, counts, seed, mha_out_w, mha_out_b,
                                        ffn_w1, ffn_b1, ffn_w2, ffn_b2,
                                        n1_g, n1_b, n2_g, n2_b,
                                        pred_w1, pred_b1, pred_w2, pred_b2, (float*)d_out);
}
